// MultiQueryAttention_41747082117701
// MI455X (gfx1250) — matrix operand order repaired and checked by reading, not compiled, not run on hardware
//
#include <hip/hip_runtime.h>

// ---------------- shapes (fixed by reference) ----------------
#define BB  2
#define SS  2048
#define HH  2048
#define HD  128
#define NH  16          // HH / HD

typedef __attribute__((ext_vector_type(16))) __bf16 v16bf;
typedef __attribute__((ext_vector_type(8)))  __bf16 v8bf;
typedef __attribute__((ext_vector_type(8)))  float  v8f;

__device__ __forceinline__ v16bf cat8(v8bf lo, v8bf hi) {
    return __builtin_shufflevector(lo, hi, 0,1,2,3,4,5,6,7,8,9,10,11,12,13,14,15);
}
__device__ __forceinline__ float redmax16(float v) {
    v = fmaxf(v, __shfl_xor(v, 1, 32));
    v = fmaxf(v, __shfl_xor(v, 2, 32));
    v = fmaxf(v, __shfl_xor(v, 4, 32));
    v = fmaxf(v, __shfl_xor(v, 8, 32));
    return v;
}
__device__ __forceinline__ float redsum16(float v) {
    v += __shfl_xor(v, 1, 32);
    v += __shfl_xor(v, 2, 32);
    v += __shfl_xor(v, 4, 32);
    v += __shfl_xor(v, 8, 32);
    return v;
}
// async global->LDS copy of 16 bytes (per lane), tracked by ASYNCcnt
__device__ __forceinline__ void async_cp16(const __bf16* g, __bf16* l) {
    unsigned lds_off = (unsigned)(uintptr_t)l;               // LDS aperture: low 32 bits
    unsigned long long ga = (unsigned long long)(uintptr_t)g;
    asm volatile("global_load_async_to_lds_b128 %0, %1, off"
                 :: "v"(lds_off), "v"(ga) : "memory");
}

// ---------------- fp32 -> bf16 conversion ----------------
__global__ void cvt_f32_bf16(const float* __restrict__ in, __bf16* __restrict__ out, int n) {
    int i = blockIdx.x * blockDim.x + threadIdx.x;
    int stride = gridDim.x * blockDim.x;
    for (; i < n; i += stride) out[i] = (__bf16)in[i];
}

// ---------------- generic NT GEMM via WMMA bf16 ----------------
// C[M,N] = A[M,K] @ W[N,K]^T + bias[N]; one wave = 32x64 tile
// mode 0: store bf16 row-major (M,N)
// mode 1: store bf16 transposed per batch: out[b*N*S + n*S + s], row = b*S + s
// mode 2: store fp32 row-major (M,N)
__global__ __launch_bounds__(256)
void gemm_nt_bf16(const __bf16* __restrict__ A, const __bf16* __restrict__ W,
                  const float* __restrict__ bias,
                  float* __restrict__ outF, __bf16* __restrict__ outB,
                  int M, int N, int K, int mode)
{
    const int wid    = (blockIdx.x * blockDim.x + threadIdx.x) >> 5;
    const int lane   = threadIdx.x & 31;
    const int half   = lane >> 4;
    const int ln     = lane & 15;
    const int mtiles = M >> 5;             // 32-row tiles
    const int mi = wid % mtiles;
    const int nj = wid / mtiles;           // 64-wide N group
    if (nj * 64 >= N) return;

    const __bf16* arow0 = A + (size_t)(mi * 32 + ln) * K;
    const __bf16* arow1 = A + (size_t)(mi * 32 + 16 + ln) * K;

    const v8f vzero = {0.f,0.f,0.f,0.f,0.f,0.f,0.f,0.f};
    v8f acc[2][4];
#pragma unroll
    for (int t = 0; t < 2; ++t)
#pragma unroll
        for (int j = 0; j < 4; ++j) acc[t][j] = vzero;

    for (int ks = 0; ks < K; ks += 32) {
        v16bf av0 = cat8(*(const v8bf*)(arow0 + ks + half * 16),
                         *(const v8bf*)(arow0 + ks + 8 + half * 16));
        v16bf av1 = cat8(*(const v8bf*)(arow1 + ks + half * 16),
                         *(const v8bf*)(arow1 + ks + 8 + half * 16));
#pragma unroll
        for (int j = 0; j < 4; ++j) {
            int n = (nj * 4 + j) * 16 + ln;
            v16bf bv = *(const v16bf*)(W + (size_t)n * K + ks + half * 16);
            acc[0][j] = __builtin_amdgcn_wmma_f32_16x16x32_bf16(
                            false, av0, false, bv, (short)0, acc[0][j], false, false);
            acc[1][j] = __builtin_amdgcn_wmma_f32_16x16x32_bf16(
                            false, av1, false, bv, (short)0, acc[1][j], false, false);
        }
    }

#pragma unroll
    for (int t = 0; t < 2; ++t)
#pragma unroll
    for (int j = 0; j < 4; ++j) {
        const int n  = (nj * 4 + j) * 16 + ln;
        const float bsv = bias ? bias[n] : 0.f;
#pragma unroll
        for (int r = 0; r < 8; ++r) {
            const int row = mi * 32 + t * 16 + r + 8 * half;
            const float v = acc[t][j][r] + bsv;
            if (mode == 2) {
                outF[(size_t)row * N + n] = v;
            } else if (mode == 0) {
                outB[(size_t)row * N + n] = (__bf16)v;
            } else {
                const int b = row / SS, s = row % SS;
                outB[(size_t)b * N * SS + (size_t)n * SS + s] = (__bf16)v;
            }
        }
    }
}

// ---------------- fused flash MQA attention ----------------
// Q: (B,S,H) bf16   K: (B,S,HD) bf16   Vt: (B,HD,S) bf16   mask: (B,1,1,S) int
// AO: (B,S,H) bf16
// Block = 8 waves, 8 consecutive 16-query tiles of the same (b,h).
// K/V tiles for each 32-key block are staged once per workgroup via async LDS copies.
__global__ __launch_bounds__(256)
void mqa_flash(const __bf16* __restrict__ Q, const __bf16* __restrict__ Kb,
               const __bf16* __restrict__ Vt, const int* __restrict__ mask,
               __bf16* __restrict__ AO)
{
    __shared__ __align__(32) __bf16 ldsK[32 * HD];       // 8KB: 32 keys x 128 d
    __shared__ __align__(32) __bf16 ldsV[HD * 32];       // 8KB: 128 d x 32 keys (V^T rows)
    __shared__ __align__(32) __bf16 ldsP[8][16 * 32];    // 8KB: per-wave P transpose

    const int tid   = threadIdx.x;
    const int wslot = tid >> 5;
    const int lane  = tid & 31;
    const int gwave = blockIdx.x * 8 + wslot;            // grid sized exactly

    const int qt = gwave % (SS / 16);
    const int bh = gwave / (SS / 16);
    const int h  = bh % NH;
    const int b  = bh / NH;

    const int half = lane >> 4;
    const int ln   = lane & 15;
    const int m    = qt * 16 + ln;

    // ---- load Q tile as 4 A-operands (hd = 4 x 32) ----
    const __bf16* qrow = Q + ((size_t)b * SS + m) * HH + h * HD;
    v16bf aq[4];
#pragma unroll
    for (int kk = 0; kk < 4; ++kk) {
        aq[kk] = cat8(*(const v8bf*)(qrow + kk * 32 + half * 16),
                      *(const v8bf*)(qrow + kk * 32 + 8 + half * 16));
    }

    const __bf16* kbase = Kb + (size_t)b * SS * HD;
    const __bf16* vbase = Vt + (size_t)b * HD * SS;
    const int*    mrow  = mask + b * SS;

    const v8f vzero = {0.f,0.f,0.f,0.f,0.f,0.f,0.f,0.f};
    v8f   Oacc[8];
    float mstat[8], lstat[8];
#pragma unroll
    for (int dc = 0; dc < 8; ++dc) Oacc[dc] = vzero;
#pragma unroll
    for (int r = 0; r < 8; ++r) { mstat[r] = -1e30f; lstat[r] = 0.f; }

    const float scale = 0.08838834764831845f;   // 1/sqrt(128)
    __bf16* lp = &ldsP[wslot][0];

    for (int kb = 0; kb < SS; kb += 32) {
        // ---- stage K (32x128) and V^T (128x32) tiles: async global->LDS ----
#pragma unroll
        for (int c = tid; c < 512; c += 256) {            // K: 512 x 16B
            const int row = c >> 4, col = (c & 15) * 8;
            async_cp16(kbase + (size_t)(kb + row) * HD + col, &ldsK[row * HD + col]);
        }
#pragma unroll
        for (int c = tid; c < 512; c += 256) {            // V: 512 x 16B
            const int d = c >> 2, col = (c & 3) * 8;
            async_cp16(vbase + (size_t)d * SS + kb + col, &ldsV[d * 32 + col]);
        }
        asm volatile("s_wait_asynccnt 0x0" ::: "memory");
        __syncthreads();

        // ---- scores: 2 x (16x16) tiles over 32 keys, K operands from LDS ----
        v8f c0 = vzero, c1 = vzero;
#pragma unroll
        for (int kk = 0; kk < 4; ++kk) {
            v16bf bv0 = *(const v16bf*)(&ldsK[(0 * 16 + ln) * HD + kk * 32 + half * 16]);
            v16bf bv1 = *(const v16bf*)(&ldsK[(1 * 16 + ln) * HD + kk * 32 + half * 16]);
            c0 = __builtin_amdgcn_wmma_f32_16x16x32_bf16(
                     false, aq[kk], false, bv0, (short)0, c0, false, false);
            c1 = __builtin_amdgcn_wmma_f32_16x16x32_bf16(
                     false, aq[kk], false, bv1, (short)0, c1, false, false);
        }

        // ---- scale + mask ----
        const int mv0 = mrow[kb + ln];
        const int mv1 = mrow[kb + 16 + ln];
#pragma unroll
        for (int r = 0; r < 8; ++r) {
            float v0 = c0[r] * scale;
            float v1 = c1[r] * scale;
            if (mv0 == 0) v0 = -1e30f;
            if (mv1 == 0) v1 = -1e30f;
            c0[r] = v0; c1[r] = v1;
        }

        // ---- online softmax stats (row-wise across 16 lanes per half) ----
        float alphar[8];
#pragma unroll
        for (int r = 0; r < 8; ++r) {
            float t = redmax16(fmaxf(c0[r], c1[r]));
            float mn = fmaxf(mstat[r], t);
            float p0 = __expf(c0[r] - mn);
            float p1 = __expf(c1[r] - mn);
            c0[r] = p0; c1[r] = p1;
            float s = redsum16(p0 + p1);
            float alpha = __expf(mstat[r] - mn);
            lstat[r] = lstat[r] * alpha + s;
            mstat[r] = mn;
            alphar[r] = alpha;
        }
#pragma unroll
        for (int dc = 0; dc < 8; ++dc)
#pragma unroll
            for (int r = 0; r < 8; ++r) Oacc[dc][r] *= alphar[r];

        // ---- transpose P (D-layout -> A-layout) through per-wave LDS slice ----
#pragma unroll
        for (int r = 0; r < 8; ++r) {
            const int row = r + 8 * half;
            lp[row * 32 + ln]      = (__bf16)c0[r];
            lp[row * 32 + 16 + ln] = (__bf16)c1[r];
        }
        asm volatile("s_wait_dscnt 0" ::: "memory");

        v16bf pa = cat8(*(const v8bf*)(lp + ln * 32 + half * 16),
                        *(const v8bf*)(lp + ln * 32 + 8 + half * 16));

        // ---- O += P @ V  (V^T rows from LDS, contiguous along keys) ----
#pragma unroll
        for (int dc = 0; dc < 8; ++dc) {
            const int d = dc * 16 + ln;
            v16bf vv = *(const v16bf*)(&ldsV[d * 32 + half * 16]);
            Oacc[dc] = __builtin_amdgcn_wmma_f32_16x16x32_bf16(
                           false, pa, false, vv, (short)0, Oacc[dc], false, false);
        }
        __syncthreads();    // protect ldsK/ldsV before next staging round
    }

    // ---- epilogue: O / l, store bf16 ----
    float inv[8];
#pragma unroll
    for (int r = 0; r < 8; ++r) inv[r] = 1.0f / lstat[r];
#pragma unroll
    for (int dc = 0; dc < 8; ++dc) {
        const int d = dc * 16 + ln;
#pragma unroll
        for (int r = 0; r < 8; ++r) {
            const int row = qt * 16 + r + 8 * half;
            AO[((size_t)b * SS + row) * HH + h * HD + d] = (__bf16)(Oacc[dc][r] * inv[r]);
        }
    }
}

// ---------------- launcher ----------------
extern "C" void kernel_launch(void* const* d_in, const int* in_sizes, int n_in,
                              void* d_out, int out_size, void* d_ws, size_t ws_size,
                              hipStream_t stream) {
    const float* hidden = (const float*)d_in[0];
    const int*   amask  = (const int*)  d_in[1];
    const float* Wq = (const float*)d_in[2];
    const float* bq = (const float*)d_in[3];
    const float* Wk = (const float*)d_in[4];
    const float* bk = (const float*)d_in[5];
    const float* Wv = (const float*)d_in[6];
    const float* bv = (const float*)d_in[7];
    const float* Wo = (const float*)d_in[8];
    const float* bo = (const float*)d_in[9];
    float* out = (float*)d_out;

    // ---- workspace carve (256B aligned) ----
    char* p = (char*)d_ws;
    auto carve = [&](size_t bytes) {
        char* r = p;
        p += (bytes + 255) & ~(size_t)255;
        return r;
    };
    __bf16* hb  = (__bf16*)carve((size_t)BB * SS * HH * 2);   // hidden bf16
    __bf16* wqb = (__bf16*)carve((size_t)HH * HH * 2);
    __bf16* wkb = (__bf16*)carve((size_t)HD * HH * 2);
    __bf16* wvb = (__bf16*)carve((size_t)HD * HH * 2);
    __bf16* wob = (__bf16*)carve((size_t)HH * HH * 2);
    __bf16* qb  = (__bf16*)carve((size_t)BB * SS * HH * 2);   // Q  (B,S,H)
    __bf16* kbf = (__bf16*)carve((size_t)BB * SS * HD * 2);   // K  (B,S,HD)
    __bf16* vtb = (__bf16*)carve((size_t)BB * HD * SS * 2);   // V^T (B,HD,S)
    __bf16* aob = (__bf16*)carve((size_t)BB * SS * HH * 2);   // attn out bf16

    // ---- conversions ----
    cvt_f32_bf16<<<2048, 256, 0, stream>>>(hidden, hb,  BB * SS * HH);
    cvt_f32_bf16<<<2048, 256, 0, stream>>>(Wq,     wqb, HH * HH);
    cvt_f32_bf16<<<256,  256, 0, stream>>>(Wk,     wkb, HD * HH);
    cvt_f32_bf16<<<256,  256, 0, stream>>>(Wv,     wvb, HD * HH);
    cvt_f32_bf16<<<2048, 256, 0, stream>>>(Wo,     wob, HH * HH);

    const int M = BB * SS;            // 4096
    // waves = (M/32) * (N/64); 8 waves per block
    auto blocks_for = [](int Mm, int Nn) { return ((Mm / 32) * (Nn / 64) + 7) / 8; };

    // ---- projections ----
    gemm_nt_bf16<<<blocks_for(M, HH), 256, 0, stream>>>(hb, wqb, bq, nullptr, qb,
                                                        M, HH, HH, 0);
    gemm_nt_bf16<<<blocks_for(M, HD), 256, 0, stream>>>(hb, wkb, bk, nullptr, kbf,
                                                        M, HD, HH, 0);
    gemm_nt_bf16<<<blocks_for(M, HD), 256, 0, stream>>>(hb, wvb, bv, nullptr, vtb,
                                                        M, HD, HH, 1);

    // ---- fused flash attention: 1 wave per 16-query tile, 8 waves/block ----
    const int attn_waves = BB * NH * (SS / 16);    // 4096
    mqa_flash<<<attn_waves / 8, 256, 0, stream>>>(qb, kbf, vtb, amask, aob);

    // ---- output projection (fp32 result) ----
    gemm_nt_bf16<<<blocks_for(M, HH), 256, 0, stream>>>(aob, wob, bo, out, nullptr,
                                                        M, HH, HH, 2);
}
